// GroupLinearBlock_24670292148733
// MI455X (gfx1250) — matrix operand order repaired and checked by reading, not compiled, not run on hardware
//
#include <hip/hip_runtime.h>

typedef _Float16 v16h __attribute__((ext_vector_type(16)));
typedef _Float16 h4   __attribute__((ext_vector_type(4)));
typedef float    v8f  __attribute__((ext_vector_type(8)));

#define N_TOK   65536
#define K_IN    1024
#define NHEAD   16
#define BN      256      // tokens per workgroup
#define KSTEPS  32       // 1024 / 32

// ---------------------------------------------------------------------------
// Prep: convert w_lin (fp32 [1024][1024]) into f16, pre-swizzled into WMMA
// B-fragment order.  Layout: i = (((kt*16 + hd)*64 + cl)*2 + h)*16 + e
//   value = w_lin[(kt*32 + h*16 + e)*1024 + hd*64 + cl]
// ---------------------------------------------------------------------------
__global__ __launch_bounds__(256) void prep_w_kernel(const float* __restrict__ w,
                                                     _Float16* __restrict__ wf) {
    int i  = blockIdx.x * 256 + threadIdx.x;   // 0 .. 2^20-1
    int e  = i & 15;
    int h  = (i >> 4) & 1;
    int cl = (i >> 5) & 63;
    int hd = (i >> 11) & 15;
    int kt = i >> 15;
    int col = hd * 64 + cl;
    int k   = kt * 32 + h * 16 + e;
    wf[i] = (_Float16)w[k * K_IN + col];
}

// Prep: grouped kernel fp32 [16][64][64] -> f16 B-fragments.
// i = ((((hd*2 + kc)*64 + n)*2 + h)*16 + e);  value = ker[hd][kc*32+h*16+e][n]
__global__ __launch_bounds__(256) void prep_k_kernel(const float* __restrict__ ker,
                                                     _Float16* __restrict__ kf) {
    int i  = blockIdx.x * 256 + threadIdx.x;   // 0 .. 65535
    int e  = i & 15;
    int h  = (i >> 4) & 1;
    int n  = (i >> 5) & 63;
    int kc = (i >> 11) & 1;
    int hd = i >> 12;
    int k  = kc * 32 + h * 16 + e;
    kf[i] = (_Float16)ker[(hd * 64 + k) * 64 + n];
}

// ---------------------------------------------------------------------------
// Fused main kernel: GEMM (f16 WMMA, fp32 accum) + per-head RMSNorm + bias
// blend + per-head 64x64 grouped GEMM.  Software-pipelined: global loads for
// k-step ks+1 are issued before the WMMAs of step ks; conversion + LDS store
// happen after.  W slices are copied with global_load_async_to_lds_b128.
// ---------------------------------------------------------------------------
__global__ __launch_bounds__(256) void glb_main(const float* __restrict__ x,
                                                const float* __restrict__ nbias,
                                                const _Float16* __restrict__ wfrag,
                                                const _Float16* __restrict__ kfrag,
                                                float* __restrict__ out) {
    __shared__ __align__(32) _Float16 lds_x[2 * BN * 32]; // 32 KB (reused as Y)
    __shared__ __align__(32) _Float16 lds_w[2 * 2048];    // 8 KB

    const int tid  = threadIdx.x;
    const int lane = tid & 31;
    const int wv   = tid >> 5;          // wave 0..7
    const int n0   = lane & 15;
    const int hh   = lane >> 4;         // half of wave

    const int head = blockIdx.x & 15;
    const int t0   = (blockIdx.x >> 4) * BN;

    // staging decomposition: each thread converts one 4-float chunk of 8 rows
    const int rsub = tid >> 3;          // 0..31
    const int c4   = (tid & 7) * 4;     // 0,4,...,28  (k within 32-chunk)
    const int g    = (c4 >> 3) & 1;                       // A-fragment group
    const int pos  = (c4 & 7) + ((c4 >> 4) & 1) * 8;      // position in group

    // LDS byte offsets (addrspace(3) offset = low 32 bits of pointer)
    const unsigned ldsw_base = (unsigned)(uintptr_t)(&lds_w[0]) + (unsigned)(tid * 16);

    v8f acc[2][4];
#pragma unroll
    for (int r = 0; r < 2; ++r)
#pragma unroll
        for (int ct = 0; ct < 4; ++ct)
#pragma unroll
            for (int v = 0; v < 8; ++v) acc[r][ct][v] = 0.0f;

    float4 xr[8];

    auto issue_loads = [&](int ks, int buf) {
        const int kbase = ks * 32;
        // x tile: 8 x float4 per thread, kept in VGPRs until after compute
#pragma unroll
        for (int p = 0; p < 8; ++p) {
            const int row = p * 32 + rsub;
            xr[p] = *(const float4*)(x + (size_t)(t0 + row) * K_IN + kbase + c4);
        }
        // W slice: async copy straight into LDS (no conversion needed)
        const unsigned long long gaddr = (unsigned long long)(uintptr_t)
            ((const char*)wfrag + (size_t)(ks * 16 + head) * 4096 + tid * 16);
        const unsigned ldsoff = ldsw_base + (unsigned)(buf * 4096);
        asm volatile("global_load_async_to_lds_b128 %0, %1, off"
                     :: "v"(ldsoff), "v"(gaddr) : "memory");
    };

    auto store_x = [&](int buf) {
        _Float16* xs = lds_x + buf * (BN * 32);
#pragma unroll
        for (int p = 0; p < 8; ++p) {
            const int row = p * 32 + rsub;
            h4 hv;
            hv[0] = (_Float16)xr[p].x; hv[1] = (_Float16)xr[p].y;
            hv[2] = (_Float16)xr[p].z; hv[3] = (_Float16)xr[p].w;
            *(h4*)(xs + row * 32 + c4) = hv;
        }
    };

    // prologue: stage k-step 0 into buffer 0
    issue_loads(0, 0);
    store_x(0);

    for (int ks = 0; ks < KSTEPS; ++ks) {
        asm volatile("s_wait_asynccnt 0x0" ::: "memory");
        __syncthreads();

        const bool more = (ks + 1) < KSTEPS;
        if (more) issue_loads(ks + 1, (ks + 1) & 1);   // loads in flight...

        const int buf = ks & 1;
        const _Float16* ws = lds_w + buf * 2048;
        const _Float16* xs = lds_x + buf * (BN * 32);

        v16h bfr[4];
#pragma unroll
        for (int ct = 0; ct < 4; ++ct)
            bfr[ct] = *(const v16h*)(ws + ((ct * 16 + n0) * 2 + hh) * 16);

#pragma unroll
        for (int r = 0; r < 2; ++r) {                  // ...under the WMMAs
            const v16h a = *(const v16h*)(xs + (wv * 32 + r * 16 + n0) * 32 + hh * 16);
#pragma unroll
            for (int ct = 0; ct < 4; ++ct)
                acc[r][ct] = __builtin_amdgcn_wmma_f32_16x16x32_f16(
                    false, a, false, bfr[ct], (short)0, acc[r][ct], false, false);
        }

        if (more) store_x((ks + 1) & 1);               // cvt + LDS store last
    }
    __syncthreads();   // all compute reads of lds_x done before Y reuse

    // ---- fused RMSNorm + bias blend; write f16 Y in A-fragment order ----
    _Float16* ys = lds_x;                      // [BN][64] f16 = 32 KB
    const float inv_s2 = 0.70710678118654752f;
#pragma unroll
    for (int r = 0; r < 2; ++r) {
        float rms[8];
#pragma unroll
        for (int v = 0; v < 8; ++v) {
            float s = 0.f;
#pragma unroll
            for (int ct = 0; ct < 4; ++ct) { float t = acc[r][ct][v]; s += t * t; }
            s += __shfl_xor(s, 1, 32);
            s += __shfl_xor(s, 2, 32);
            s += __shfl_xor(s, 4, 32);
            s += __shfl_xor(s, 8, 32);         // stays inside 16-lane half
            rms[v] = rsqrtf(s * (1.0f / 64.0f) + 1e-6f);
        }
#pragma unroll
        for (int ct = 0; ct < 4; ++ct) {
            const int d  = ct * 16 + n0;
            const int kc = d >> 5, kl = d & 31;
            const int gg = (kl >> 3) & 1;
            const int pp = (kl & 7) + ((kl >> 4) & 1) * 8;
#pragma unroll
            for (int v = 0; v < 8; ++v) {
                const int rloc  = wv * 32 + r * 16 + v + 8 * hh;
                const int token = t0 + rloc;
                const float b = nbias[(size_t)token * 1024 + head * 64 + d];
                const float y = (acc[r][ct][v] * rms[v] + b) * inv_s2;
                ys[rloc * 64 + d] = (_Float16)y;
            }
        }
    }
    __syncthreads();

    // ---- second GEMM: [16x64] x [64x64] per head via 2 K-chunks ----
    v16h kb[2][4];
#pragma unroll
    for (int kc = 0; kc < 2; ++kc)
#pragma unroll
        for (int ft = 0; ft < 4; ++ft)
            kb[kc][ft] = *(const v16h*)(kfrag +
                (((head * 2 + kc) * 64 + ft * 16 + n0) * 2 + hh) * 16);

#pragma unroll
    for (int r = 0; r < 2; ++r) {
        v8f o[4];
#pragma unroll
        for (int ft = 0; ft < 4; ++ft)
#pragma unroll
            for (int v = 0; v < 8; ++v) o[ft][v] = 0.0f;

#pragma unroll
        for (int kc = 0; kc < 2; ++kc) {
            const v16h a = *(const v16h*)(ys +
                (wv * 32 + r * 16 + n0) * 64 + kc * 32 + hh * 16);
#pragma unroll
            for (int ft = 0; ft < 4; ++ft)
                o[ft] = __builtin_amdgcn_wmma_f32_16x16x32_f16(
                    false, a, false, kb[kc][ft], (short)0, o[ft], false, false);
        }
#pragma unroll
        for (int ft = 0; ft < 4; ++ft) {
            const int col = head * 64 + ft * 16 + n0;
#pragma unroll
            for (int v = 0; v < 8; ++v) {
                const int token = t0 + wv * 32 + r * 16 + v + 8 * hh;
                out[(size_t)token * 1024 + col] = o[ft][v];
            }
        }
    }
}

// ---------------------------------------------------------------------------
extern "C" void kernel_launch(void* const* d_in, const int* in_sizes, int n_in,
                              void* d_out, int out_size, void* d_ws, size_t ws_size,
                              hipStream_t stream) {
    const float* x  = (const float*)d_in[0];   // [65536,1024]
    const float* nb = (const float*)d_in[1];   // [65536,16,64]
    const float* wl = (const float*)d_in[2];   // [1024,1024]
    const float* kr = (const float*)d_in[3];   // [16,64,64]

    _Float16* wf = (_Float16*)d_ws;                          // 2 MB
    _Float16* kf = (_Float16*)((char*)d_ws + (1u << 21));    // 128 KB

    prep_w_kernel<<<(K_IN * K_IN) / 256, 256, 0, stream>>>(wl, wf);
    prep_k_kernel<<<(NHEAD * 2 * 64 * 32) / 256, 256, 0, stream>>>(kr, kf);
    glb_main<<<(N_TOK / BN) * NHEAD, 256, 0, stream>>>(x, nb, wf, kf, (float*)d_out);
}
